// LanguageModel_76682346103174
// MI455X (gfx1250) — matrix operand order repaired and checked by reading, not compiled, not run on hardware
//
#include <hip/hip_runtime.h>
#include <hip/hip_fp16.h>
#include <hip/hip_bf16.h>

typedef __attribute__((ext_vector_type(16))) _Float16 v16h;
typedef __attribute__((ext_vector_type(8)))  _Float16 v8h;
typedef __attribute__((ext_vector_type(4)))  _Float16 v4h;
typedef __attribute__((ext_vector_type(8)))  float    v8f;
typedef __attribute__((ext_vector_type(4)))  unsigned v4u;
typedef __attribute__((ext_vector_type(8)))  unsigned v8u;

#define VOCAB 32000
#define HID   1024
#define SEQ   128
#define BAT   32
#define G4    (4 * HID)     // 4096 gate columns
#define ROWS  (SEQ * BAT)   // 4096
#define NBLK  32            // persistent blocks for LSTM scan

#define WMMA_F16(a, b, c) \
    __builtin_amdgcn_wmma_f32_16x16x32_f16(false, (a), false, (b), (short)0, (c), false, false)

// ---------------- WMMA fragment loaders (CDNA5 16x16x32 f16 layouts) ----------------
// A: 16x32 (MxK). lane&15 = row M; hi=lane>>4 selects K chunks {hi*8..+7, 16+hi*8..+7}
__device__ __forceinline__ v16h ld_a(const _Float16* __restrict__ base, int ld, int lane) {
    int row = lane & 15, hi = lane >> 4;
    const _Float16* r = base + (size_t)row * ld + hi * 8;
    v16h a;
    *(v8h*)&a         = *(const v8h*)(r);        // K = hi*8 .. hi*8+7
    *(((v8h*)&a) + 1) = *(const v8h*)(r + 16);   // K = 16+hi*8 .. 16+hi*8+7
    return a;
}
// B: 32x16 (KxN) from row-major [N,K] weights. lane&15 = column N (= weight row);
// hi=lane>>4 selects K chunks {hi*8..+7, 16+hi*8..+7} (same as A).
__device__ __forceinline__ v16h ld_b(const _Float16* __restrict__ base, int ld, int lane) {
    int col = lane & 15, hi = lane >> 4;
    const _Float16* r = base + (size_t)col * ld + hi * 8;
    v16h b;
    *(v8h*)&b         = *(const v8h*)(r);
    *(((v8h*)&b) + 1) = *(const v8h*)(r + 16);
    return b;
}

__device__ __forceinline__ float sigf(float x) { return 1.0f / (1.0f + __expf(-x)); }

// ---------------- vectorized f32 -> f16 convert (b128 in / b64 out) ----------------
__global__ __launch_bounds__(256) void cvt_f16(const float* __restrict__ in,
                                               _Float16* __restrict__ out, size_t n4) {
    size_t i = (size_t)blockIdx.x * blockDim.x + threadIdx.x;
    size_t stride = (size_t)gridDim.x * blockDim.x;
    const float4* in4 = (const float4*)in;
    v4h* out4 = (v4h*)out;
    for (; i < n4; i += stride) {
        float4 v = in4[i];
        v4h o;
        o[0] = (_Float16)v.x; o[1] = (_Float16)v.y;
        o[2] = (_Float16)v.z; o[3] = (_Float16)v.w;
        out4[i] = o;
    }
}

// ---------------- embedding gather + convert (one row per block) ----------------
__global__ __launch_bounds__(256) void embed_f16(const int* __restrict__ x,
                                                 const float* __restrict__ emb,
                                                 _Float16* __restrict__ e) {
    int row = blockIdx.x;                       // row = s*B + b
    int idx = x[row];
    const float4* src = (const float4*)(emb + (size_t)idx * HID);
    v4h* dst = (v4h*)(e + (size_t)row * HID);
    int i = threadIdx.x;                        // 256 threads x 4 floats = 1024
    float4 v = src[i];
    v4h o;
    o[0] = (_Float16)v.x; o[1] = (_Float16)v.y;
    o[2] = (_Float16)v.z; o[3] = (_Float16)v.w;
    dst[i] = o;
}

// ---------------- tiled f16 WMMA GEMM: C[M,N] = A[M,K] * B[N,K]^T + bias0 + bias1 ----
// 256 threads = 8 waves; block tile 64x128; wave (wv>>2) -> M half, (wv&3) -> N quarter.
// Each wave: 32x32 register tile = 4 accumulators; K-loop software-pipelined so the
// next K-tile's 8 b128 loads are in flight while 4 WMMAs execute on the current one.
__global__ __launch_bounds__(256) void gemm_f16_wmma(const _Float16* __restrict__ A,
                                                     const _Float16* __restrict__ B,
                                                     const float* __restrict__ bias0,
                                                     const float* __restrict__ bias1,
                                                     float* __restrict__ C,
                                                     int K, int ldc, int streamOut) {
    const int lane = threadIdx.x & 31;
    const int wv   = threadIdx.x >> 5;
    const int mBase = blockIdx.x * 64 + (wv >> 2) * 32;
    const int nBase = blockIdx.y * 128 + (wv & 3) * 32;
    const _Float16* A0 = A + (size_t)mBase * K;
    const _Float16* A1 = A + (size_t)(mBase + 16) * K;
    const _Float16* B0 = B + (size_t)nBase * K;
    const _Float16* B1 = B + (size_t)(nBase + 16) * K;

    v8f acc00 = {}, acc01 = {}, acc10 = {}, acc11 = {};
    // prologue: fragments for k = 0
    v16h a0 = ld_a(A0, K, lane), a1 = ld_a(A1, K, lane);
    v16h b0 = ld_b(B0, K, lane), b1 = ld_b(B1, K, lane);

    for (int k = 32; k < K; k += 32) {
        if (k + 224 < K) {  // global_prefetch_b8, ~7 tiles ahead
            __builtin_prefetch(A0 + k + 224, 0, 1);
            __builtin_prefetch(A1 + k + 224, 0, 1);
            __builtin_prefetch(B0 + k + 224, 0, 1);
            __builtin_prefetch(B1 + k + 224, 0, 1);
        }
        // issue next K-tile loads before consuming current fragments
        v16h na0 = ld_a(A0 + k, K, lane);
        v16h na1 = ld_a(A1 + k, K, lane);
        v16h nb0 = ld_b(B0 + k, K, lane);
        v16h nb1 = ld_b(B1 + k, K, lane);
        acc00 = WMMA_F16(a0, b0, acc00);
        acc01 = WMMA_F16(a0, b1, acc01);
        acc10 = WMMA_F16(a1, b0, acc10);
        acc11 = WMMA_F16(a1, b1, acc11);
        a0 = na0; a1 = na1; b0 = nb0; b1 = nb1;
    }
    acc00 = WMMA_F16(a0, b0, acc00);
    acc01 = WMMA_F16(a0, b1, acc01);
    acc10 = WMMA_F16(a1, b0, acc10);
    acc11 = WMMA_F16(a1, b1, acc11);

    const int col = lane & 15, hi = lane >> 4;
    const int c0 = nBase + col, c1 = nBase + 16 + col;
    float bb0 = (bias0 ? bias0[c0] : 0.0f) + (bias1 ? bias1[c0] : 0.0f);
    float bb1 = (bias0 ? bias0[c1] : 0.0f) + (bias1 ? bias1[c1] : 0.0f);
#pragma unroll
    for (int r = 0; r < 8; ++r) {
        size_t row0 = (size_t)(mBase + hi * 8 + r);
        size_t row1 = row0 + 16;
        float v00 = acc00[r] + bb0, v01 = acc01[r] + bb1;
        float v10 = acc10[r] + bb0, v11 = acc11[r] + bb1;
        if (streamOut) {  // 524 MB logits stream: non-temporal
            __builtin_nontemporal_store(v00, &C[row0 * ldc + c0]);
            __builtin_nontemporal_store(v01, &C[row0 * ldc + c1]);
            __builtin_nontemporal_store(v10, &C[row1 * ldc + c0]);
            __builtin_nontemporal_store(v11, &C[row1 * ldc + c1]);
        } else {
            C[row0 * ldc + c0] = v00;
            C[row0 * ldc + c1] = v01;
            C[row1 * ldc + c0] = v10;
            C[row1 * ldc + c1] = v11;
        }
    }
}

// ---------------- counter reset for grid barrier ----------------
__global__ void zero2(unsigned* p) {
    if (threadIdx.x == 0) { p[0] = 0u; p[1] = 0u; }
}

// ---------------- persistent LSTM scan ----------------
// 32 blocks; block j owns H-slice [j*32, j*32+32). Its 128 gate columns
// {g*H + hbase + jj : g=0..3, jj=0..31} of Whh live in LDS (256 KB of the 320 KB WGP LDS),
// staged by the Tensor Data Mover (4x 2D TENSOR_LOAD_TO_LDS, one per gate).
// Per step: G[32b x 128col] = h[32x1024] x WhhSlice^T via WMMA, gates fused, h ping-pong.
__global__ __launch_bounds__(256) void lstm_scan(const float* __restrict__ gx,      // [S*B, 4H]
                                                 const _Float16* __restrict__ Whh,  // [4H, H] f16
                                                 const float* __restrict__ h0,
                                                 const float* __restrict__ c0,
                                                 _Float16* __restrict__ hb0,
                                                 _Float16* __restrict__ hb1,
                                                 _Float16* __restrict__ y,          // [S*B, H] f16
                                                 float* __restrict__ hN,
                                                 float* __restrict__ cN,
                                                 unsigned* __restrict__ bar) {
    extern __shared__ char smem[];
    _Float16* Wl = (_Float16*)smem;                                    // [128][1024] 256 KB
    float* G  = (float*)(smem + (size_t)128 * HID * 2);                // [32][128]   16 KB
    float* Cs = (float*)(smem + (size_t)128 * HID * 2 + 32 * 128 * 4); // [32][32]     4 KB

    const int tid = threadIdx.x, lane = tid & 31, wv = tid >> 5;
    const int hbase = blockIdx.x * 32;

    // ---- TDM: stage Whh slice into LDS. Gate g occupies LDS rows [g*32, g*32+32),
    // sourced from Whh rows [g*H + hbase, +32), row stride H, 2-byte elements. ----
    if (wv == 0) {
        unsigned ldsbase = (unsigned)(size_t)(void*)Wl;
#pragma unroll
        for (int g = 0; g < 4; ++g) {
            unsigned long long ga =
                (unsigned long long)(size_t)&Whh[((size_t)(g * HID + hbase)) * HID];
            v4u d0;
            d0[0] = 1u;                                        // count=1 valid descriptor
            d0[1] = ldsbase + (unsigned)(g * 32 * HID * 2);    // lds_addr (bytes)
            d0[2] = (unsigned)(ga & 0xFFFFFFFFu);              // global_addr[31:0]
            d0[3] = (unsigned)((ga >> 32) & 0x01FFFFFFu)       // global_addr[56:32]
                    | (2u << 30);                              // type=2 ("image")
            v8u d1;
            d1[0] = (1u << 16);                 // wg_mask=0, data_size=1 (2 bytes)
            d1[1] = ((unsigned)HID & 0xFFFFu) << 16;           // tensor_dim0[15:0]
            d1[2] = ((unsigned)HID >> 16) | ((unsigned)G4 << 16); // td0[31:16] | td1[15:0]
            d1[3] = ((unsigned)G4 >> 16) | ((unsigned)HID << 16); // td1[31:16] | tile_dim0
            d1[4] = 32u;                        // tile_dim1=32 rows, tile_dim2=0
            d1[5] = (unsigned)HID;              // tensor_dim0_stride[31:0]
            d1[6] = 0u;                         // stride0[47:32] | stride1[15:0]
            d1[7] = 0u;                         // tensor_dim1_stride[47:16]
            asm volatile("tensor_load_to_lds %0, %1" :: "s"(d0), "s"(d1) : "memory");
        }
#if __has_builtin(__builtin_amdgcn_s_wait_tensorcnt)
        __builtin_amdgcn_s_wait_tensorcnt(0);
#else
        asm volatile("s_wait_tensorcnt 0x0" ::: "memory");
#endif
    }
    // Init h (global ping-pong buffer, own columns only) and c (LDS)
    for (int i = tid; i < BAT * 32; i += 256) {
        int b = i >> 5, jj = i & 31;
        hb0[b * HID + hbase + jj] = (_Float16)h0[b * HID + hbase + jj];
        Cs[i] = c0[b * HID + hbase + jj];
    }
    __syncthreads();   // Wl (TDM) + Cs visible to all waves

    unsigned my_gen = 0;
    for (int t = 0; t < SEQ; ++t) {
        // ---- grid-wide barrier: all h writes of step t-1 (or init) visible ----
        __syncthreads();
        __threadfence();
        if (tid == 0) {
            unsigned target = ++my_gen;
            unsigned old = atomicAdd(&bar[0], 1u);
            if (old == NBLK - 1) {
                atomicExch(&bar[0], 0u);
                atomicAdd(&bar[1], 1u);
            } else {
                while (atomicAdd(&bar[1], 0u) < target) __builtin_amdgcn_s_sleep(1);
            }
        }
        __syncthreads();
        __threadfence();

        const _Float16* hr = (t & 1) ? hb1 : hb0;
        _Float16*       hw = (t & 1) ? hb0 : hb1;

        // ---- G = h_prev x WhhSlice^T : M=32, N=128, K=1024; wave wv owns N-tile wv*16 ----
        // software-pipelined: next K-tile (1 ds B-frag + 2 global A-frags) in flight
        v8f acc0 = {}, acc1 = {};
        const _Float16* Bp = &Wl[(size_t)(wv * 16) * HID];
        v16h bf  = ld_b(Bp, HID, lane);
        v16h af0 = ld_a(hr, HID, lane);              // batches 0..15
        v16h af1 = ld_a(hr + 16 * HID, HID, lane);   // batches 16..31
        for (int k = 32; k < HID; k += 32) {
            v16h nbf  = ld_b(Bp + k, HID, lane);
            v16h naf0 = ld_a(hr + k, HID, lane);
            v16h naf1 = ld_a(hr + 16 * HID + k, HID, lane);
            acc0 = WMMA_F16(af0, bf, acc0);
            acc1 = WMMA_F16(af1, bf, acc1);
            bf = nbf; af0 = naf0; af1 = naf1;
        }
        acc0 = WMMA_F16(af0, bf, acc0);
        acc1 = WMMA_F16(af1, bf, acc1);
        {
            int col = lane & 15, hi = lane >> 4;
#pragma unroll
            for (int r = 0; r < 8; ++r) {
                G[(hi * 8 + r) * 128 + wv * 16 + col]      = acc0[r];
                G[(16 + hi * 8 + r) * 128 + wv * 16 + col] = acc1[r];
            }
        }
        __syncthreads();

        // ---- fused gates: c' = sig(f)*c + sig(i)*tanh(g); h' = sig(o)*tanh(c') ----
        for (int i = tid; i < BAT * 32; i += 256) {
            int b = i >> 5, jj = i & 31;
            const float* gr = gx + ((size_t)t * BAT + b) * G4 + hbase + jj;
            float xi = gr[0]       + G[b * 128 +      jj];
            float xf = gr[HID]     + G[b * 128 + 32 + jj];
            float xg = gr[2 * HID] + G[b * 128 + 64 + jj];
            float xo = gr[3 * HID] + G[b * 128 + 96 + jj];
            float cv = sigf(xf) * Cs[i] + sigf(xi) * tanhf(xg);
            float hv = sigf(xo) * tanhf(cv);
            Cs[i] = cv;
            _Float16 hh = (_Float16)hv;
            hw[b * HID + hbase + jj] = hh;
            y[((size_t)t * BAT + b) * HID + hbase + jj] = hh;
            if (t == SEQ - 1) {
                hN[b * HID + hbase + jj] = hv;
                cN[b * HID + hbase + jj] = cv;
            }
        }
    }
}

// ---------------- host driver ----------------
extern "C" void kernel_launch(void* const* d_in, const int* in_sizes, int n_in,
                              void* d_out, int out_size, void* d_ws, size_t ws_size,
                              hipStream_t stream) {
    (void)in_sizes; (void)n_in; (void)out_size; (void)ws_size;
    const int*   x    = (const int*)d_in[0];
    const float* h0_1 = (const float*)d_in[1];
    const float* c0_1 = (const float*)d_in[2];
    const float* h0_2 = (const float*)d_in[3];
    const float* c0_2 = (const float*)d_in[4];
    const float* emb  = (const float*)d_in[5];
    const float* Wih1 = (const float*)d_in[6];
    const float* Whh1 = (const float*)d_in[7];
    const float* bih1 = (const float*)d_in[8];
    const float* bhh1 = (const float*)d_in[9];
    const float* Wih2 = (const float*)d_in[10];
    const float* Whh2 = (const float*)d_in[11];
    const float* bih2 = (const float*)d_in[12];
    const float* bhh2 = (const float*)d_in[13];
    const float* Wlin = (const float*)d_in[14];
    const float* blin = (const float*)d_in[15];
    float* out = (float*)d_out;

    // workspace carve (256B aligned)
    char* ws = (char*)d_ws;
    size_t off = 0;
    auto carve = [&](size_t bytes) -> void* {
        void* p = (void*)(ws + off);
        off = (off + bytes + 255) & ~(size_t)255;
        return p;
    };
    unsigned* bar   = (unsigned*)carve(256);
    _Float16* eF    = (_Float16*)carve((size_t)ROWS * HID * 2);
    _Float16* W1iF  = (_Float16*)carve((size_t)G4 * HID * 2);
    _Float16* W1hF  = (_Float16*)carve((size_t)G4 * HID * 2);
    _Float16* W2iF  = (_Float16*)carve((size_t)G4 * HID * 2);
    _Float16* W2hF  = (_Float16*)carve((size_t)G4 * HID * 2);
    _Float16* WlinF = (_Float16*)carve((size_t)VOCAB * HID * 2);
    float*    gxbuf = (float*)   carve((size_t)ROWS * G4 * 4);
    _Float16* y1F   = (_Float16*)carve((size_t)ROWS * HID * 2);
    _Float16* y2F   = (_Float16*)carve((size_t)ROWS * HID * 2);
    _Float16* hb0   = (_Float16*)carve((size_t)BAT * HID * 2);
    _Float16* hb1   = (_Float16*)carve((size_t)BAT * HID * 2);

    const size_t LOGITS = (size_t)ROWS * VOCAB;
    float* h1o = out + LOGITS;
    float* c1o = h1o + (size_t)BAT * HID;
    float* h2o = c1o + (size_t)BAT * HID;
    float* c2o = h2o + (size_t)BAT * HID;

    const int SMEM = 128 * HID * 2 + 32 * 128 * 4 + 32 * 32 * 4;  // 282,624 B < 320 KB WGP LDS
    hipFuncSetAttribute((const void*)lstm_scan,
                        hipFuncAttributeMaxDynamicSharedMemorySize, SMEM);

    // 1) weight conversions to f16 (vectorized, counts are elements/4)
    cvt_f16<<<512, 256, 0, stream>>>(Wih1, W1iF, (size_t)G4 * HID / 4);
    cvt_f16<<<512, 256, 0, stream>>>(Whh1, W1hF, (size_t)G4 * HID / 4);
    cvt_f16<<<512, 256, 0, stream>>>(Wih2, W2iF, (size_t)G4 * HID / 4);
    cvt_f16<<<512, 256, 0, stream>>>(Whh2, W2hF, (size_t)G4 * HID / 4);
    cvt_f16<<<2048, 256, 0, stream>>>(Wlin, WlinF, (size_t)VOCAB * HID / 4);

    // 2) embedding gather
    embed_f16<<<ROWS, 256, 0, stream>>>(x, emb, eF);

    // 3) layer-1 input GEMM: gx = e * Wih1^T + bih1 + bhh1
    gemm_f16_wmma<<<dim3(ROWS / 64, G4 / 128), 256, 0, stream>>>(
        eF, W1iF, bih1, bhh1, gxbuf, HID, G4, 0);

    // 4) layer-1 recurrence (persistent, grid-barrier per step, Whh in LDS via TDM)
    zero2<<<1, 32, 0, stream>>>(bar);
    lstm_scan<<<NBLK, 256, SMEM, stream>>>(gxbuf, W1hF, h0_1, c0_1, hb0, hb1,
                                           y1F, h1o, c1o, bar);

    // 5) layer-2 input GEMM: gx = y1 * Wih2^T + bih2 + bhh2
    gemm_f16_wmma<<<dim3(ROWS / 64, G4 / 128), 256, 0, stream>>>(
        y1F, W2iF, bih2, bhh2, gxbuf, HID, G4, 0);

    // 6) layer-2 recurrence
    zero2<<<1, 32, 0, stream>>>(bar);
    lstm_scan<<<NBLK, 256, SMEM, stream>>>(gxbuf, W2hF, h0_2, c0_2, hb0, hb1,
                                           y2F, h2o, c2o, bar);

    // 7) output projection: logits = y2 * Wlin^T + blin (268 GFLOP, streamed f32 out)
    gemm_f16_wmma<<<dim3(ROWS / 64, VOCAB / 128), 256, 0, stream>>>(
        y2F, WlinF, blin, nullptr, out, HID, VOCAB, 1);
}
